// Attention_55688545960525
// MI455X (gfx1250) — matrix operand order repaired and checked by reading, not compiled, not run on hardware
//
#include <hip/hip_runtime.h>
#include <hip/hip_bf16.h>
#include <stdint.h>

// ---------------------------------------------------------------------------
// CDNA5 (gfx1250) bf16 WMMA attention: qkv GEMM -> flash attention (TDM-fed)
// -> output projection.
// ---------------------------------------------------------------------------

typedef __attribute__((ext_vector_type(16))) __bf16 v16bf;
typedef __attribute__((ext_vector_type(2)))  __bf16 v2bf;
typedef __attribute__((ext_vector_type(8)))  float  v8f;
typedef __attribute__((ext_vector_type(4)))  uint32_t u32x4;
typedef __attribute__((ext_vector_type(8)))  uint32_t u32x8;

union FragBF {
    v16bf v;
    uint4 q[2];
    unsigned short us[16];
};

__device__ __forceinline__ unsigned short f2bf_bits(float f) {
    union { float f; uint32_t u; } v; v.f = f;
    uint32_t r = v.u + 0x7FFFu + ((v.u >> 16) & 1u);
    return (unsigned short)(r >> 16);
}

// packed f32x2 -> bf16x2 (single v_cvt_pk_bf16_f32 when available)
__device__ __forceinline__ uint32_t f2bf_pk(float lo, float hi) {
#if __has_builtin(__builtin_amdgcn_cvt_pk_bf16_f32)
    v2bf r = __builtin_amdgcn_cvt_pk_bf16_f32(lo, hi);
    uint32_t u; __builtin_memcpy(&u, &r, 4);
    return u;
#else
    return (uint32_t)f2bf_bits(lo) | ((uint32_t)f2bf_bits(hi) << 16);
#endif
}

__device__ __forceinline__ v8f vzero8() {
    v8f z;
#pragma unroll
    for (int i = 0; i < 8; ++i) z[i] = 0.0f;
    return z;
}

__device__ __forceinline__ v8f wmma_bf16(const FragBF& a, const FragBF& b, v8f c) {
    return __builtin_amdgcn_wmma_f32_16x16x32_bf16(
        false, a.v, false, b.v, (short)0, c, false, false);
}

// Tensor Data Mover: contiguous 1-D copy of n_elems bf16 from global -> LDS.
__device__ __forceinline__ void tdm_load_lds_1d(const unsigned short* gsrc,
                                                uint32_t lds_off_bytes,
                                                uint32_t n_elems) {
    uint64_t ga = (uint64_t)(uintptr_t)gsrc;
    u32x4 g0;
    g0[0] = 1u;                                           // count=1 valid descriptor
    g0[1] = lds_off_bytes;                                // lds_addr
    g0[2] = (uint32_t)ga;                                 // global_addr[31:0]
    g0[3] = (uint32_t)((ga >> 32) & 0x01FFFFFFu) | (2u << 30);  // addr[56:32] | type=2
    u32x8 g1;
    g1[0] = 0x00010000u;                                  // data_size=2B
    g1[1] = (n_elems & 0xFFFFu) << 16;                    // tensor_dim0[15:0]
    g1[2] = ((n_elems >> 16) & 0xFFFFu) | (1u << 16);     // tensor_dim0[31:16] | tensor_dim1=1
    g1[3] = (n_elems & 0xFFFFu) << 16;                    // tile_dim0
    g1[4] = 0u;
    g1[5] = n_elems;                                      // tensor_dim0_stride lo
    g1[6] = 0u;
    g1[7] = 0u;
    asm volatile("tensor_load_to_lds %0, %1" :: "s"(g0), "s"(g1) : "memory");
}

// ===========================================================================
// Kernel 1: qkv = x @ qkv_w ; scatter into q (scaled), k, v  [B,H,N,64] bf16
// ===========================================================================
__global__ __launch_bounds__(256)
void attn_qkv_gemm_gfx1250(const float* __restrict__ x,
                           const float* __restrict__ w,
                           unsigned short* __restrict__ qws,
                           unsigned short* __restrict__ kws,
                           unsigned short* __restrict__ vws)
{
    __shared__ unsigned short As[128][40];  // [m][k]
    __shared__ unsigned short Bs[128][40];  // transposed [n][k]

    const int t    = threadIdx.x;
    const int lane = t & 31;
    const int wv   = t >> 5;
    const int hi   = lane >> 4;
    const int ln   = lane & 15;
    const int row0 = blockIdx.y * 128;
    const int col0 = blockIdx.x * 128;
    const int mb   = (wv >> 1) * 32;
    const int nb   = (wv & 1) * 64;

    v8f acc[2][4];
#pragma unroll
    for (int i = 0; i < 2; ++i)
#pragma unroll
        for (int j = 0; j < 4; ++j) acc[i][j] = vzero8();

    const int arow = t >> 1, acol = (t & 1) * 16;
    const int bk   = t >> 3, bnof = (t & 7) * 16;

    for (int k0 = 0; k0 < 1024; k0 += 32) {
        {   // A tile: packed converts, 32-bit LDS stores
            const float4* src = (const float4*)(x + (size_t)(row0 + arow) * 1024 + k0 + acol);
#pragma unroll
            for (int j = 0; j < 4; ++j) {
                float4 f = src[j];
                *(uint32_t*)&As[arow][acol + j*4 + 0] = f2bf_pk(f.x, f.y);
                *(uint32_t*)&As[arow][acol + j*4 + 2] = f2bf_pk(f.z, f.w);
            }
        }
        {   // B tile transposed: packed converts, scattered b16 stores
            const float4* src = (const float4*)(w + (size_t)(k0 + bk) * 3072 + col0 + bnof);
#pragma unroll
            for (int j = 0; j < 4; ++j) {
                float4 f = src[j];
                uint32_t p0 = f2bf_pk(f.x, f.y);
                uint32_t p1 = f2bf_pk(f.z, f.w);
                Bs[bnof + j*4 + 0][bk] = (unsigned short)p0;
                Bs[bnof + j*4 + 1][bk] = (unsigned short)(p0 >> 16);
                Bs[bnof + j*4 + 2][bk] = (unsigned short)p1;
                Bs[bnof + j*4 + 3][bk] = (unsigned short)(p1 >> 16);
            }
        }
        __syncthreads();

        FragBF a[2], b[4];
#pragma unroll
        for (int i = 0; i < 2; ++i) {
            int r  = mb + i * 16 + ln;
            int kh = hi * 8;
            a[i].q[0] = *(const uint4*)&As[r][kh];
            a[i].q[1] = *(const uint4*)&As[r][kh + 16];
        }
#pragma unroll
        for (int j = 0; j < 4; ++j) {
            int n  = nb + j * 16 + ln;
            int kh = hi * 8;
            b[j].q[0] = *(const uint4*)&Bs[n][kh];
            b[j].q[1] = *(const uint4*)&Bs[n][kh + 16];
        }
#pragma unroll
        for (int i = 0; i < 2; ++i)
#pragma unroll
            for (int j = 0; j < 4; ++j)
                acc[i][j] = wmma_bf16(a[i], b[j], acc[i][j]);
        __syncthreads();
    }

    // block-uniform q/k/v select
    const int which = col0 >> 10;
    unsigned short* __restrict__ dst = (which == 0) ? qws : (which == 1 ? kws : vws);
    const float scl = (which == 0) ? 0.125f : 1.0f;
    const int cb = (col0 & 1023) + nb;

#pragma unroll
    for (int i = 0; i < 2; ++i) {
#pragma unroll
        for (int j = 0; j < 4; ++j) {
            const int c = cb + j * 16 + ln;
            const int h = c >> 6, d = c & 63;
#pragma unroll
            for (int e = 0; e < 8; e += 2) {
                const uint32_t pk = f2bf_pk(acc[i][j][e] * scl, acc[i][j][e + 1] * scl);
                const int g0r = row0 + mb + i * 16 + e + hi * 8;
                const int g1r = g0r + 1;
                dst[(((size_t)(g0r >> 10) * 16 + h) * 1024 + (g0r & 1023)) * 64 + d] =
                    (unsigned short)pk;
                dst[(((size_t)(g1r >> 10) * 16 + h) * 1024 + (g1r & 1023)) * 64 + d] =
                    (unsigned short)(pk >> 16);
            }
        }
    }
}

// ===========================================================================
// Kernel 2: flash attention. WG = 256 thr (8 waves x 16 rows), 64-key tiles
// DMA'd into LDS by the Tensor Data Mover, K frags from LDS, V transposed
// LDS->LDS, P through LDS, online softmax via shfl_xor + v_exp_f32.
// ===========================================================================
__global__ __launch_bounds__(256)
void attn_flash_gfx1250(const unsigned short* __restrict__ qws,
                        const unsigned short* __restrict__ kws,
                        const unsigned short* __restrict__ vws,
                        const float* __restrict__ atts,
                        unsigned short* __restrict__ ows)
{
    __shared__ unsigned short Kraw[64 * 64];
    __shared__ unsigned short Vraw[64 * 64];
    __shared__ unsigned short Vt[64][72];
    __shared__ unsigned short Pt[8][16][72];

    const int t    = threadIdx.x;
    const int lane = t & 31;
    const int wv   = t >> 5;
    const int hi   = lane >> 4;
    const int ln   = lane & 15;
    const int b    = blockIdx.z;
    const int h    = blockIdx.y;
    const int row0 = blockIdx.x * 128 + wv * 16;

    const size_t head_off = ((size_t)b * 16 + h) * 1024 * 64;
    const unsigned short* qh  = qws + head_off;
    const unsigned short* kh_ = kws + head_off;
    const unsigned short* vh  = vws + head_off;

    FragBF qa[2];
#pragma unroll
    for (int f = 0; f < 2; ++f) {
        const unsigned short* p = qh + (size_t)(row0 + ln) * 64 + f * 32 + hi * 8;
        qa[f].q[0] = *(const uint4*)p;
        qa[f].q[1] = *(const uint4*)(p + 16);
    }

    v8f acc[4];
#pragma unroll
    for (int i = 0; i < 4; ++i) acc[i] = vzero8();
    v8f mrow, lrow;
#pragma unroll
    for (int e = 0; e < 8; ++e) { mrow[e] = -__builtin_inff(); lrow[e] = 0.0f; }

    const uint32_t kraw_off = (uint32_t)(uintptr_t)&Kraw[0];
    const uint32_t vraw_off = (uint32_t)(uintptr_t)&Vraw[0];
    const int vkey = t >> 2, vd0 = (t & 3) * 16;

    for (int kt = 0; kt < 16; ++kt) {
        const int key0 = kt * 64;
        __syncthreads();
        if (wv == 0) {
            tdm_load_lds_1d(kh_ + (size_t)key0 * 64, kraw_off, 4096);
            tdm_load_lds_1d(vh + (size_t)key0 * 64, vraw_off, 4096);
            __builtin_amdgcn_s_wait_tensorcnt(0);
        }
        __syncthreads();

        {   // Vraw[key][d] -> Vt[d][key]
            FragBF vtmp;
            const uint4* src = (const uint4*)&Vraw[vkey * 64 + vd0];
            vtmp.q[0] = src[0];
            vtmp.q[1] = src[1];
#pragma unroll
            for (int j = 0; j < 16; ++j) Vt[vd0 + j][vkey] = vtmp.us[j];
        }
        __syncthreads();

        v8f s[4];
#pragma unroll
        for (int i = 0; i < 4; ++i) {
            const unsigned short* p0 = &Kraw[(i * 16 + ln) * 64 + hi * 8];
            FragBF kb0, kb1;
            kb0.q[0] = *(const uint4*)p0;
            kb0.q[1] = *(const uint4*)(p0 + 16);
            kb1.q[0] = *(const uint4*)(p0 + 32);
            kb1.q[1] = *(const uint4*)(p0 + 48);
            v8f z = vzero8();
            z = wmma_bf16(qa[0], kb0, z);
            z = wmma_bf16(qa[1], kb1, z);
            s[i] = z;
        }

        const float* ab = atts + ((size_t)b * 1024 + row0) * 1024 + key0;
#pragma unroll
        for (int i = 0; i < 4; ++i)
#pragma unroll
            for (int e = 0; e < 8; ++e)
                s[i][e] += ab[(size_t)(e + hi * 8) * 1024 + i * 16 + ln];

        v8f cmax = s[0];
#pragma unroll
        for (int i = 1; i < 4; ++i)
#pragma unroll
            for (int e = 0; e < 8; ++e) cmax[e] = fmaxf(cmax[e], s[i][e]);
#pragma unroll
        for (int off = 1; off < 16; off <<= 1)
#pragma unroll
            for (int e = 0; e < 8; ++e)
                cmax[e] = fmaxf(cmax[e], __shfl_xor(cmax[e], off, 32));

        v8f mnew, scale;
#pragma unroll
        for (int e = 0; e < 8; ++e) {
            mnew[e]  = fmaxf(mrow[e], cmax[e]);
            scale[e] = __expf(mrow[e] - mnew[e]);
        }
#pragma unroll
        for (int i = 0; i < 4; ++i)
#pragma unroll
            for (int e = 0; e < 8; ++e) acc[i][e] *= scale[e];

        v8f rsum = vzero8();
#pragma unroll
        for (int i = 0; i < 4; ++i)
#pragma unroll
            for (int e = 0; e < 8; ++e) {
                float p = __expf(s[i][e] - mnew[e]);
                s[i][e] = p;
                rsum[e] += p;
            }
#pragma unroll
        for (int off = 1; off < 16; off <<= 1)
#pragma unroll
            for (int e = 0; e < 8; ++e)
                rsum[e] += __shfl_xor(rsum[e], off, 32);
#pragma unroll
        for (int e = 0; e < 8; ++e) {
            lrow[e] = lrow[e] * scale[e] + rsum[e];
            mrow[e] = mnew[e];
        }

        // P (D-layout) -> LDS, packed converts
#pragma unroll
        for (int i = 0; i < 4; ++i) {
#pragma unroll
            for (int e = 0; e < 8; e += 2) {
                uint32_t pk = f2bf_pk(s[i][e], s[i][e + 1]);
                Pt[wv][e + hi * 8][i * 16 + ln]     = (unsigned short)pk;
                Pt[wv][e + 1 + hi * 8][i * 16 + ln] = (unsigned short)(pk >> 16);
            }
        }

        FragBF pa[2];
#pragma unroll
        for (int f = 0; f < 2; ++f) {
            pa[f].q[0] = *(const uint4*)&Pt[wv][ln][f * 32 + hi * 8];
            pa[f].q[1] = *(const uint4*)&Pt[wv][ln][f * 32 + hi * 8 + 16];
        }
#pragma unroll
        for (int i = 0; i < 4; ++i) {
#pragma unroll
            for (int f = 0; f < 2; ++f) {
                FragBF vb;
                vb.q[0] = *(const uint4*)&Vt[i * 16 + ln][f * 32 + hi * 8];
                vb.q[1] = *(const uint4*)&Vt[i * 16 + ln][f * 32 + hi * 8 + 16];
                acc[i] = wmma_bf16(pa[f], vb, acc[i]);
            }
        }
    }

    // normalize (8 divides hoisted) + packed-converted stores
    v8f invl;
#pragma unroll
    for (int e = 0; e < 8; ++e) invl[e] = 1.0f / lrow[e];
#pragma unroll
    for (int i = 0; i < 4; ++i) {
        const int col = h * 64 + i * 16 + ln;
#pragma unroll
        for (int e = 0; e < 8; e += 2) {
            uint32_t pk = f2bf_pk(acc[i][e] * invl[e], acc[i][e + 1] * invl[e + 1]);
            size_t r0 = (size_t)b * 1024 + row0 + e + hi * 8;
            ows[r0 * 1024 + col]       = (unsigned short)pk;
            ows[(r0 + 1) * 1024 + col] = (unsigned short)(pk >> 16);
        }
    }
}

// ===========================================================================
// Kernel 3: out = attn @ proj_w + proj_b
// ===========================================================================
__global__ __launch_bounds__(256)
void attn_proj_gemm_gfx1250(const unsigned short* __restrict__ attn,
                            const float* __restrict__ w,
                            const float* __restrict__ bias,
                            float* __restrict__ out)
{
    __shared__ unsigned short Ws[128][40];

    const int t    = threadIdx.x;
    const int lane = t & 31;
    const int wv   = t >> 5;
    const int hi   = lane >> 4;
    const int ln   = lane & 15;
    const int row0 = blockIdx.y * 128;
    const int col0 = blockIdx.x * 128;
    const int mb   = (wv >> 1) * 32;
    const int nb   = (wv & 1) * 64;

    v8f acc[2][4];
#pragma unroll
    for (int i = 0; i < 2; ++i)
#pragma unroll
        for (int j = 0; j < 4; ++j) acc[i][j] = vzero8();

    const int bk = t >> 3, bnof = (t & 7) * 16;

    for (int k0 = 0; k0 < 1024; k0 += 32) {
        {
            const float4* src = (const float4*)(w + (size_t)(k0 + bk) * 1024 + col0 + bnof);
#pragma unroll
            for (int j = 0; j < 4; ++j) {
                float4 f = src[j];
                uint32_t p0 = f2bf_pk(f.x, f.y);
                uint32_t p1 = f2bf_pk(f.z, f.w);
                Ws[bnof + j*4 + 0][bk] = (unsigned short)p0;
                Ws[bnof + j*4 + 1][bk] = (unsigned short)(p0 >> 16);
                Ws[bnof + j*4 + 2][bk] = (unsigned short)p1;
                Ws[bnof + j*4 + 3][bk] = (unsigned short)(p1 >> 16);
            }
        }
        __syncthreads();

        FragBF a[2], b[4];
#pragma unroll
        for (int i = 0; i < 2; ++i) {
            const unsigned short* p =
                attn + (size_t)(row0 + mb + i * 16 + ln) * 1024 + k0 + hi * 8;
            a[i].q[0] = *(const uint4*)p;
            a[i].q[1] = *(const uint4*)(p + 16);
        }
#pragma unroll
        for (int j = 0; j < 4; ++j) {
            int n  = nb + j * 16 + ln;
            int kh = hi * 8;
            b[j].q[0] = *(const uint4*)&Ws[n][kh];
            b[j].q[1] = *(const uint4*)&Ws[n][kh + 16];
        }
#pragma unroll
        for (int i = 0; i < 2; ++i)
#pragma unroll
            for (int j = 0; j < 4; ++j)
                acc[i][j] = wmma_bf16(a[i], b[j], acc[i][j]);
        __syncthreads();
    }

#pragma unroll
    for (int i = 0; i < 2; ++i) {
#pragma unroll
        for (int j = 0; j < 4; ++j) {
            int gcol = col0 + nb + j * 16 + ln;
            float bv = bias[gcol];
#pragma unroll
            for (int e = 0; e < 8; ++e) {
                int grow = row0 + mb + i * 16 + e + hi * 8;
                out[(size_t)grow * 1024 + gcol] = acc[i][j][e] + bv;
            }
        }
    }
}

// ===========================================================================
extern "C" void kernel_launch(void* const* d_in, const int* in_sizes, int n_in,
                              void* d_out, int out_size, void* d_ws, size_t ws_size,
                              hipStream_t stream) {
    const float* x      = (const float*)d_in[0];
    // d_in[1] = text_embeds: dead code in reference, unused
    const float* atts   = (const float*)d_in[2];
    const float* qkv_w  = (const float*)d_in[3];
    const float* proj_w = (const float*)d_in[4];
    const float* proj_b = (const float*)d_in[5];
    float* out = (float*)d_out;

    const size_t HE = (size_t)8 * 16 * 1024 * 64;
    unsigned short* qws  = (unsigned short*)d_ws;
    unsigned short* kws  = qws + HE;
    unsigned short* vws  = qws + 2 * HE;
    unsigned short* aout = qws + 3 * HE;

    attn_qkv_gemm_gfx1250<<<dim3(24, 64), 256, 0, stream>>>(x, qkv_w, qws, kws, vws);
    attn_flash_gfx1250<<<dim3(8, 16, 8), 256, 0, stream>>>(qws, kws, vws, atts, aout);
    attn_proj_gemm_gfx1250<<<dim3(8, 64), 256, 0, stream>>>(aout, proj_w, proj_b, out);
}
